// GNNRecommender_82918638617247
// MI455X (gfx1250) — matrix operand order repaired and checked by reading, not compiled, not run on hardware
//
#include <hip/hip_runtime.h>
#include <hip/hip_bf16.h>

// ---------------------------------------------------------------------------
// Types for CDNA5 WMMA
// ---------------------------------------------------------------------------
typedef __attribute__((ext_vector_type(16))) __bf16          v16bf;
typedef __attribute__((ext_vector_type(8)))  float           v8f;
typedef __attribute__((ext_vector_type(8)))  unsigned short  us8;
typedef __attribute__((ext_vector_type(16))) unsigned short  us16;

static __device__ __forceinline__ unsigned short f2bf(float f) {
    unsigned int u = __builtin_bit_cast(unsigned int, f);
    unsigned int r = (u + 0x7FFFu + ((u >> 16) & 1u)) >> 16;   // round-to-nearest-even
    return (unsigned short)r;
}

// ---------------------------------------------------------------------------
// GEMM:  out[M x 64] (+)= A[M x K] (fp32, row-major, stride lda) * W[K x 64]
// W supplied pre-converted+transposed as WT[64][kpad] bf16 (zero padded).
// Block = 128 rows x 64 cols, 8 waves; wave w owns rows [w*16, w*16+16) and
// all 4 column tiles (4 x v8f accumulators). K split across gridDim.y with
// fp32 atomic reduction when gridDim.y > 1 (out zeroed first).
// LDS is double buffered: one barrier per K-step; next tile's global loads
// are issued before the current tile's WMMAs.
// ---------------------------------------------------------------------------
#define TILE_M 128

__global__ __launch_bounds__(256)
void gemm_bf16_wmma(const float* __restrict__ A, int lda,
                    const unsigned short* __restrict__ WT, int kpad, int K,
                    float* __restrict__ out, int M)
{
    __shared__ unsigned short a_lds[2][TILE_M * 32];  // [row][k], bf16 bits
    __shared__ unsigned short b_lds[2][64 * 32];      // [n][k],  bf16 bits

    const int tid   = threadIdx.x;
    const int lane  = tid & 31;
    const int wave  = tid >> 5;
    const int mBase = blockIdx.x * TILE_M;
    const bool fullM = (mBase + TILE_M) <= M;         // block-uniform fast path

    v8f acc0 = {}, acc1 = {}, acc2 = {}, acc3 = {};

    // A staging role: each thread owns 16 consecutive k of one row
    const int  arow = tid >> 1;
    const int  acol = (tid & 1) * 16;
    const int  grow = mBase + arow;
    const bool rok  = (grow < M);
    // B staging role: each thread copies one us8 (8 bf16)
    const int  bn = tid >> 2;
    const int  bk = (tid & 3) * 8;
    // Fragment geometry (ISA 7.12.2 layouts)
    const int frow = wave * 16 + (lane & 15);   // A-matrix row within tile
    const int fah  = (lane >> 4) * 8;           // A k-half offset (0 / 8)
    const int fbh  = (lane >> 4) * 8;           // B k-half offset (0 / 8)

    const int totalSteps = kpad >> 5;
    const int perY = (totalSteps + gridDim.y - 1) / gridDim.y;
    const int s0 = blockIdx.y * perY;
    const int s1 = (s0 + perY < totalSteps) ? (s0 + perY) : totalSteps;

    // ---- tile staging (A: fp32 global -> bf16 LDS, B: bf16 global -> LDS) ----
    auto stage = [&](int s, int buf) {
        const int kb = s << 5;
        const int k0 = kb + acol;
        unsigned short* dstp = &a_lds[buf][arow * 32 + acol];
        if (fullM && (k0 + 16) <= K) {              // uniform for interior blocks
            const float4* src4 = (const float4*)&A[(long)grow * lda + k0];
            us8 pk0, pk1;
            {
                float4 v0 = src4[0], v1 = src4[1];
                pk0[0] = f2bf(v0.x); pk0[1] = f2bf(v0.y);
                pk0[2] = f2bf(v0.z); pk0[3] = f2bf(v0.w);
                pk0[4] = f2bf(v1.x); pk0[5] = f2bf(v1.y);
                pk0[6] = f2bf(v1.z); pk0[7] = f2bf(v1.w);
            }
            {
                float4 v2 = src4[2], v3 = src4[3];
                pk1[0] = f2bf(v2.x); pk1[1] = f2bf(v2.y);
                pk1[2] = f2bf(v2.z); pk1[3] = f2bf(v2.w);
                pk1[4] = f2bf(v3.x); pk1[5] = f2bf(v3.y);
                pk1[6] = f2bf(v3.z); pk1[7] = f2bf(v3.w);
            }
            *(us8*)(dstp)     = pk0;                // ds_store_b128
            *(us8*)(dstp + 8) = pk1;                // ds_store_b128
            if (kb + 64 < kpad)                     // stream-ahead hint on x
                __builtin_prefetch(&A[(long)grow * lda + k0 + 64], 0, 1);
        } else {
            #pragma unroll
            for (int j = 0; j < 16; ++j) {
                const int k = k0 + j;
                float v = (rok && k < K) ? A[(long)grow * lda + k] : 0.0f;
                dstp[j] = f2bf(v);
            }
        }
        // B tile (kpad is zero padded, always in range)
        us8 bv = *(const us8*)&WT[(long)bn * kpad + kb + bk];
        *(us8*)&b_lds[buf][bn * 32 + bk] = bv;
    };

    if (s0 < s1) stage(s0, 0);                      // prologue

    for (int s = s0; s < s1; ++s) {
        const int p = (s - s0) & 1;
        __syncthreads();                            // tile p staged; buf p^1 free
        if (s + 1 < s1) stage(s + 1, p ^ 1);        // overlap next loads w/ math

        // ---- A fragment (two b128 LDS reads) ----
        us16 ar;
        {
            us8 lo = *(const us8*)&a_lds[p][frow * 32 + fah];
            us8 hi = *(const us8*)&a_lds[p][frow * 32 + fah + 16];
            #pragma unroll
            for (int j = 0; j < 8; ++j) { ar[j] = lo[j]; ar[j + 8] = hi[j]; }
        }
        const v16bf af = __builtin_bit_cast(v16bf, ar);

        // ---- all 4 B fragments, then 4 back-to-back WMMAs ----
        us16 br[4];
        #pragma unroll
        for (int ct = 0; ct < 4; ++ct) {
            const int n = ct * 16 + (lane & 15);
            us8 lo = *(const us8*)&b_lds[p][n * 32 + fbh];
            us8 hi = *(const us8*)&b_lds[p][n * 32 + fbh + 16];
            #pragma unroll
            for (int j = 0; j < 8; ++j) { br[ct][j] = lo[j]; br[ct][j + 8] = hi[j]; }
        }
        acc0 = __builtin_amdgcn_wmma_f32_16x16x32_bf16(false, af, false,
                   __builtin_bit_cast(v16bf, br[0]), (short)0, acc0, false, false);
        acc1 = __builtin_amdgcn_wmma_f32_16x16x32_bf16(false, af, false,
                   __builtin_bit_cast(v16bf, br[1]), (short)0, acc1, false, false);
        acc2 = __builtin_amdgcn_wmma_f32_16x16x32_bf16(false, af, false,
                   __builtin_bit_cast(v16bf, br[2]), (short)0, acc2, false, false);
        acc3 = __builtin_amdgcn_wmma_f32_16x16x32_bf16(false, af, false,
                   __builtin_bit_cast(v16bf, br[3]), (short)0, acc3, false, false);
    }

    // ---- store / accumulate D (VGPR r -> row r, +8 for upper lanes) ----
    const int col0  = lane & 15;
    const int rbase = mBase + wave * 16 + (lane >> 4) * 8;
    if (gridDim.y == 1) {
        #pragma unroll
        for (int rr = 0; rr < 8; ++rr) {
            const int row = rbase + rr;
            if (row < M) {
                out[(long)row * 64 + col0 +  0] = acc0[rr];
                out[(long)row * 64 + col0 + 16] = acc1[rr];
                out[(long)row * 64 + col0 + 32] = acc2[rr];
                out[(long)row * 64 + col0 + 48] = acc3[rr];
            }
        }
    } else {
        #pragma unroll
        for (int rr = 0; rr < 8; ++rr) {
            const int row = rbase + rr;
            if (row < M) {
                atomicAdd(&out[(long)row * 64 + col0 +  0], acc0[rr]);
                atomicAdd(&out[(long)row * 64 + col0 + 16], acc1[rr]);
                atomicAdd(&out[(long)row * 64 + col0 + 32], acc2[rr]);
                atomicAdd(&out[(long)row * 64 + col0 + 48], acc3[rr]);
            }
        }
    }
}

// ---------------------------------------------------------------------------
// Small helper kernels
// ---------------------------------------------------------------------------
__global__ void zero_f32(float* p, int n) {
    int i = blockIdx.x * blockDim.x + threadIdx.x;
    if (i < n) p[i] = 0.0f;
}

__global__ void deg_init(float* deg, int n) {       // self-loop counts as 1
    int i = blockIdx.x * blockDim.x + threadIdx.x;
    if (i < n) deg[i] = 1.0f;
}

__global__ void deg_count(const long long* __restrict__ dst, float* deg, int E) {
    int e = blockIdx.x * blockDim.x + threadIdx.x;
    if (e < E) atomicAdd(&deg[(int)dst[e]], 1.0f);
}

__global__ void deg_inv(float* deg, int n) {
    int i = blockIdx.x * blockDim.x + threadIdx.x;
    if (i < n) { float d = deg[i]; deg[i] = (d > 0.0f) ? rsqrtf(d) : 0.0f; }
}

// W [K x 64] fp32 -> WT [64 x kpad] bf16, zero padded in k
__global__ void convert_WT(const float* __restrict__ W, unsigned short* __restrict__ WT,
                           int K, int kpad) {
    int idx = blockIdx.x * blockDim.x + threadIdx.x;
    if (idx >= 64 * kpad) return;
    int n = idx / kpad, k = idx % kpad;
    WT[idx] = (k < K) ? f2bf(W[k * 64 + n]) : (unsigned short)0;
}

// z[i][f] = h[i][f] * dinv[i]^2   (self-loop term initializes the aggregate)
__global__ void selfloop_init(const float* __restrict__ h, const float* __restrict__ dinv,
                              float* __restrict__ z, int total) {
    int t = blockIdx.x * blockDim.x + threadIdx.x;
    if (t < total) { float di = dinv[t >> 6]; z[t] = h[t] * di * di; }
}

// z[dst][f] += h[src][f] * dinv[src]*dinv[dst]  for every edge
__global__ void scatter_edges(const float* __restrict__ h, const float* __restrict__ dinv,
                              const long long* __restrict__ src, const long long* __restrict__ dst,
                              float* z, long long total) {
    long long t = (long long)blockIdx.x * blockDim.x + threadIdx.x;
    if (t >= total) return;
    int e = (int)(t >> 6), f = (int)(t & 63);
    int s = (int)src[e], d = (int)dst[e];
    float nrm = dinv[s] * dinv[d];
    atomicAdd(&z[d * 64 + f], h[s * 64 + f] * nrm);
}

__global__ void bias_relu(const float* __restrict__ z, const float* __restrict__ b,
                          float* __restrict__ o, int total) {
    int t = blockIdx.x * blockDim.x + threadIdx.x;
    if (t < total) o[t] = fmaxf(z[t] + b[t & 63], 0.0f);
}

__global__ void bias_add(const float* __restrict__ z, const float* __restrict__ b,
                         float* __restrict__ o, int total) {
    int t = blockIdx.x * blockDim.x + threadIdx.x;
    if (t < total) o[t] = z[t] + b[t & 63];
}

// ---------------------------------------------------------------------------
// Driver
// ---------------------------------------------------------------------------
extern "C" void kernel_launch(void* const* d_in, const int* in_sizes, int n_in,
                              void* d_out, int out_size, void* d_ws, size_t ws_size,
                              hipStream_t stream) {
    const float*     x  = (const float*)d_in[0];      // [N, N] fp32
    const long long* ei = (const long long*)d_in[1];  // [2, E] int64
    const float*     W1 = (const float*)d_in[2];      // [N, 64]
    const float*     b1 = (const float*)d_in[3];      // [64]
    const float*     W2 = (const float*)d_in[4];      // [64, 64]
    const float*     b2 = (const float*)d_in[5];      // [64]
    float*           out = (float*)d_out;             // [N, 64]

    const int N = in_sizes[2] / 64;
    const int E = in_sizes[1] / 2;
    const int Kpad1 = ((N + 31) / 32) * 32;

    // workspace carve-out
    char*  ws  = (char*)d_ws;
    size_t off = 0;
    auto take = [&](size_t bytes) -> char* {
        char* p = ws + off;
        off += (bytes + 255) & ~(size_t)255;
        return p;
    };
    float*          dinv = (float*)take((size_t)N * 4);
    unsigned short* W1T  = (unsigned short*)take((size_t)64 * Kpad1 * 2);
    unsigned short* W2T  = (unsigned short*)take((size_t)64 * 64 * 2);
    float*          h1   = (float*)take((size_t)N * 64 * 4);
    float*          z1   = (float*)take((size_t)N * 64 * 4);
    float*          h2   = (float*)take((size_t)N * 64 * 4);
    float*          z2   = (float*)take((size_t)N * 64 * 4);

    const int NT = 256;
    const int NF = N * 64;
    const long long EF = (long long)E * 64;
    const long long* esrc = ei;
    const long long* edst = ei + E;

    // --- normalization coefficients ---
    deg_init <<<(N + NT - 1) / NT, NT, 0, stream>>>(dinv, N);
    deg_count<<<(E + NT - 1) / NT, NT, 0, stream>>>(edst, dinv, E);
    deg_inv  <<<(N + NT - 1) / NT, NT, 0, stream>>>(dinv, N);

    // --- weight conversion (fp32 -> transposed bf16) ---
    convert_WT<<<(64 * Kpad1 + NT - 1) / NT, NT, 0, stream>>>(W1, W1T, N, Kpad1);
    convert_WT<<<(64 * 64   + NT - 1) / NT, NT, 0, stream>>>(W2, W2T, 64, 64);

    // --- layer 1: h1 = x @ W1 (bf16 WMMA, K split 8-way) ---
    zero_f32<<<(NF + NT - 1) / NT, NT, 0, stream>>>(h1, NF);
    dim3 g1((N + TILE_M - 1) / TILE_M, 8);
    gemm_bf16_wmma<<<g1, 256, 0, stream>>>(x, N, W1T, Kpad1, N, h1, N);

    // --- layer 1 aggregation + bias + relu (result into h1's buffer) ---
    selfloop_init<<<(NF + NT - 1) / NT, NT, 0, stream>>>(h1, dinv, z1, NF);
    scatter_edges<<<(unsigned)((EF + NT - 1) / NT), NT, 0, stream>>>(h1, dinv, esrc, edst, z1, EF);
    bias_relu<<<(NF + NT - 1) / NT, NT, 0, stream>>>(z1, b1, h1, NF);

    // --- layer 2: h2 = relu_out @ W2 ---
    dim3 g2((N + TILE_M - 1) / TILE_M, 1);
    gemm_bf16_wmma<<<g2, 256, 0, stream>>>(h1, 64, W2T, 64, 64, h2, N);

    // --- layer 2 aggregation + bias ---
    selfloop_init<<<(NF + NT - 1) / NT, NT, 0, stream>>>(h2, dinv, z2, NF);
    scatter_edges<<<(unsigned)((EF + NT - 1) / NT), NT, 0, stream>>>(h2, dinv, esrc, edst, z2, EF);
    bias_add<<<(NF + NT - 1) / NT, NT, 0, stream>>>(z2, b2, out, NF);
}
